// BiLSTMCRF_4973572129077
// MI455X (gfx1250) — matrix operand order repaired and checked by reading, not compiled, not run on hardware
//
#include <hip/hip_runtime.h>
#include <hip/hip_bf16.h>

// BiLSTM-CRF for MI455X (gfx1250, wave32).
// - All large GEMMs via v_wmma_f32_16x16x32_bf16 (f32 accumulation).
// - Shared weight tiles staged in LDS once per block via the Tensor Data
//   Mover (tensor_load_to_lds + s_wait_tensorcnt), with TDM LDS padding used
//   to de-conflict banks. Fallback cooperative copy if builtin unavailable.
// - Sequential LSTM recurrence = 256 fused step kernels; W_hh is L2-resident.

#define V_ 50000
#define E_ 512
#define H_ 1024
#define T_ 21
#define B_ 64
#define L_ 256
#define G4H (4 * H_)     // 4096
#define M1 (L_ * B_)     // 16384
#define K2H (2 * H_)     // 2048

// padded LDS row strides (bytes): row + 4 dwords pad per 256 data dwords
#define XP_LDSTRIDE 1040   // 512 bf16 = 256 dw -> 260 dw  (bank shift 4, conflict-free)
#define ST_LDSTRIDE 2080   // 1024 bf16 = 512 dw -> 520 dw (bank shift 8, 2-way)

typedef __attribute__((ext_vector_type(16))) __bf16 v16bf;
typedef __attribute__((ext_vector_type(8)))  float  v8f;

#if __has_builtin(__builtin_amdgcn_tensor_load_to_lds) && \
    __has_builtin(__builtin_amdgcn_s_wait_tensorcnt)
#define USE_TDM 1
#else
#define USE_TDM 0
#endif

__device__ inline __bf16 f2bf(float f) {
    union { unsigned short u; __bf16 b; } cv;
    unsigned x = __float_as_uint(f);
    cv.u = (unsigned short)((x + 0x7FFFu + ((x >> 16) & 1u)) >> 16);  // RNE
    return cv.b;
}

// ---- WMMA fragment loaders (wave32, contiguous k order in both A and B) -------------------
// A (16x32 bf16, row-major source, row baked into rowptr):
//   lanes 0-15 : elems 0..15 -> K=k0..k0+15
//   lanes 16-31: same +16
__device__ inline v16bf load_a_row(const __bf16* __restrict__ rowptr, int k0, int lane) {
    int kb = k0 + ((lane >> 4) << 4);
    const __bf16* p = rowptr + kb;
    v16bf a;
    *((uint4*)&a)     = *(const uint4*)(p);
    *((uint4*)&a + 1) = *(const uint4*)(p + 8);
    return a;
}
__device__ inline v16bf load_a(const __bf16* __restrict__ A, int ld, int m, int k0, int lane) {
    return load_a_row(A + (size_t)m * ld, k0, lane);
}
// B (32x16 bf16) from weight matrix stored row-major [N][K] in global memory
__device__ inline v16bf load_b(const __bf16* __restrict__ W, int ld, int n, int k0, int lane) {
    int kb = k0 + ((lane >> 4) << 4);
    const __bf16* p = W + (size_t)n * ld + kb;
    v16bf b;
    *((uint4*)&b)     = *(const uint4*)(p);
    *((uint4*)&b + 1) = *(const uint4*)(p + 8);
    return b;
}
// B fragment from a TDM-padded LDS tile: row-major rows of bf16, rowstride
// bytes per row, +16B pad inserted after each 256 data dwords within a row.
__device__ inline v16bf load_b_ldsp(const char* lds, int row, int k0, int lane, int rowstride) {
    int kb = k0 + ((lane >> 4) << 4);            // element index, multiple of 16
    int kdw = kb >> 1;                           // dword index, multiple of 8
    int off = row * rowstride + kdw * 4 + (kdw >= 256 ? 16 : 0);
    v16bf b;
    *((uint4*)&b)     = *(const uint4*)(lds + off);
    *((uint4*)&b + 1) = *(const uint4*)(lds + off + 16);
    return b;
}
__device__ inline v8f wmma_bf16(v16bf a, v16bf b, v8f c) {
    return __builtin_amdgcn_wmma_f32_16x16x32_bf16(false, a, false, b, (short)0, c, false, false);
}

// ---- Tensor Data Mover: 2-D tile (tileK x tileRows bf16) global -> LDS -----
#if USE_TDM
typedef __attribute__((ext_vector_type(4))) unsigned int u32x4;
typedef __attribute__((ext_vector_type(8))) int i32x8;
typedef __attribute__((ext_vector_type(4))) int i32x4;

__device__ inline unsigned lds_off32(const void* p) {
    return (unsigned)(unsigned long long)p;   // LDS aperture: low 32 bits = offset
}

// D# per CDNA5 ISA. data_size=2B; pad_enable, interval=256dw,
// amount=4dw. tensor dims == tile dims (tile is fully in-bounds).
__device__ inline void tdm_load_2d_pad(const void* gsrc, unsigned ldsOff,
                                       int tileK, int tileRows, int strideK) {
    unsigned long long ga = (unsigned long long)gsrc;
    u32x4 g0;
    g0[0] = 1u;                                               // count=1, user desc
    g0[1] = ldsOff;                                           // lds_addr
    g0[2] = (unsigned)ga;                                     // global_addr[31:0]
    g0[3] = (unsigned)((ga >> 32) & 0x1FFFFFFu) | (2u << 30); // addr[56:32] | type=2
    i32x8 g1;
    g1[0] = (int)((1u << 16) | (1u << 20) | (7u << 22) | (3u << 25)); // 2B, pad 256dw/+4dw
    g1[1] = (int)(((unsigned)tileK & 0xFFFFu) << 16);         // tensor_dim0[15:0]
    g1[2] = (int)((((unsigned)tileK >> 16) & 0xFFFFu) |
                  (((unsigned)tileRows & 0xFFFFu) << 16));    // dim0 hi | dim1 lo
    g1[3] = (int)(((unsigned)tileK & 0xFFFFu) << 16);         // dim1 hi=0 | tile_dim0
    g1[4] = (int)((unsigned)tileRows & 0xFFFFu);              // tile_dim1 | tile_dim2=0
    g1[5] = strideK;                                          // tensor_dim0_stride lo
    g1[6] = 0;
    g1[7] = 0;
    i32x4 z4 = {0, 0, 0, 0};
#if __clang_major__ >= 23
    i32x8 z8 = {0, 0, 0, 0, 0, 0, 0, 0};
    __builtin_amdgcn_tensor_load_to_lds(g0, g1, z4, z4, z8, 0);
#else
    __builtin_amdgcn_tensor_load_to_lds(g0, g1, z4, z4, 0);
#endif
}
#endif  // USE_TDM

// ---- setup kernels --------------------------------------------------------
__global__ void __launch_bounds__(256) k_cvt_bf16(const float* __restrict__ s,
                                                  __bf16* __restrict__ d, int n) {
    int i = blockIdx.x * 256 + threadIdx.x;
    if (i < n) d[i] = f2bf(s[i]);
}

__global__ void __launch_bounds__(256) k_pack_wout(const float* __restrict__ W,
                                                   __bf16* __restrict__ Wo) {
    int i = blockIdx.x * 256 + threadIdx.x;   // over 32*2048 (rows >= 21 zero-padded)
    if (i < 32 * K2H) {
        int r = i >> 11;
        Wo[i] = (r < T_) ? f2bf(W[(size_t)r * K2H + (i & (K2H - 1))]) : f2bf(0.f);
    }
}

__global__ void __launch_bounds__(256) k_bias(const float* a0, const float* a1,
                                              const float* b0, const float* b1,
                                              float* of, float* ob) {
    int i = blockIdx.x * 256 + threadIdx.x;
    if (i < G4H) { of[i] = a0[i] + a1[i]; ob[i] = b0[i] + b1[i]; }
}

// embedding gather -> bf16 x matrix, layout [m = l*B + b][E]; token 0 -> zeros
__global__ void __launch_bounds__(256) k_embed(const int* __restrict__ ids,
                                               const float* __restrict__ tab,
                                               __bf16* __restrict__ xbf) {
    int lb = blockIdx.x;                 // m = l*B + b
    int l = lb >> 6, b = lb & 63;
    int id = ids[b * L_ + l];            // input_ids is (B, L)
    const float* src = tab + (size_t)id * E_;
    __bf16* dst = xbf + (size_t)lb * E_;
    for (int e = threadIdx.x; e < E_; e += 256)
        dst[e] = (id == 0) ? f2bf(0.f) : f2bf(src[e]);
}

// zero h(-1), h(L), c states
__global__ void __launch_bounds__(256) k_init(__bf16* hsf0, __bf16* hsbL,
                                              float* cf, float* cb) {
    int i = blockIdx.x * 256 + threadIdx.x;
    if (i < B_ * H_) {
        hsf0[i] = f2bf(0.f); hsbL[i] = f2bf(0.f);
        cf[i] = 0.f; cb[i] = 0.f;
    }
}

// ---- x_proj = x @ W_ih^T + (b_ih + b_hh) : M=16384, N=4096, K=512, 2 dirs --
// block: 256 thr (8 waves); wave = 16x64 f32 tile; block = 128x64.
// The 64-col W tile (64x512 bf16) is shared by all 8 waves -> stage in LDS.
__global__ void __launch_bounds__(256) k_xproj(const __bf16* __restrict__ xbf,
                                               const __bf16* __restrict__ Wf,
                                               const __bf16* __restrict__ Wb,
                                               const float* __restrict__ bsf,
                                               const float* __restrict__ bsb,
                                               float* __restrict__ outf,
                                               float* __restrict__ outb) {
    __shared__ uint4 ldsB4[64 * XP_LDSTRIDE / 16];
    char* ldsB = (char*)ldsB4;
    const int dir = blockIdx.z;
    const __bf16* W = dir ? Wb : Wf;
    const float* bs = dir ? bsb : bsf;
    float* out      = dir ? outb : outf;
    const int lane = threadIdx.x & 31, wave = threadIdx.x >> 5;
    const int m0 = blockIdx.x * 128 + wave * 16;
    const int n0 = blockIdx.y * 64;
    const int mr = m0 + (lane & 15);
    const int nc = lane & 15;

#if USE_TDM
    if (wave == 0) {
        tdm_load_2d_pad(W + (size_t)n0 * E_, lds_off32(ldsB), E_, 64, E_);
        __builtin_amdgcn_s_wait_tensorcnt(0);
    }
#else
    for (int u = threadIdx.x; u < 64 * 64; u += 256) {  // 16B units: 64 rows x 64
        int r = u >> 6, kdw = (u & 63) * 4;             // K=512 -> kdw<256, no mid pad
        *(uint4*)(ldsB + r * XP_LDSTRIDE + kdw * 4) =
            *(const uint4*)(W + (size_t)(n0 + r) * E_ + kdw * 2);
    }
#endif
    __syncthreads();

    v8f c0 = {}, c1 = {}, c2 = {}, c3 = {};
    for (int k0 = 0; k0 < E_; k0 += 32) {
        v16bf a  = load_a(xbf, E_, mr, k0, lane);
        v16bf b0 = load_b_ldsp(ldsB,      nc, k0, lane, XP_LDSTRIDE);
        v16bf b1 = load_b_ldsp(ldsB, 16 + nc, k0, lane, XP_LDSTRIDE);
        v16bf b2 = load_b_ldsp(ldsB, 32 + nc, k0, lane, XP_LDSTRIDE);
        v16bf b3 = load_b_ldsp(ldsB, 48 + nc, k0, lane, XP_LDSTRIDE);
        c0 = wmma_bf16(a, b0, c0);
        c1 = wmma_bf16(a, b1, c1);
        c2 = wmma_bf16(a, b2, c2);
        c3 = wmma_bf16(a, b3, c3);
    }
    const int hi = lane >> 4;
    float bv0 = bs[n0 + nc], bv1 = bs[n0 + 16 + nc];
    float bv2 = bs[n0 + 32 + nc], bv3 = bs[n0 + 48 + nc];
    for (int r = 0; r < 8; r++) {
        int m = m0 + r + 8 * hi;
        float* o = out + (size_t)m * G4H + n0;
        o[nc]      = c0[r] + bv0;
        o[16 + nc] = c1[r] + bv1;
        o[32 + nc] = c2[r] + bv2;
        o[48 + nc] = c3[r] + bv3;
    }
}

// ---- fused LSTM step: g = x_proj[t] + h_{t-1} @ W_hh^T, then cell update ---
// grid (64 j-groups, 2 dirs), block 128 (4 waves). Wave w: batch rows 16w..,
// cols {g*H + jg*16 .. +15} for gates g=0..3 -> 4 C fragments.
// The 64 shared W_hh rows (4 gate chunks of 16 x 1024 bf16) staged in LDS:
// wave g DMAs gate-chunk g via TDM, then block syncs.
__global__ void __launch_bounds__(128) k_step(const float* __restrict__ xpf,
                                              const float* __restrict__ xpb,
                                              const __bf16* __restrict__ Whf,
                                              const __bf16* __restrict__ Whb,
                                              __bf16* __restrict__ hsf,
                                              __bf16* __restrict__ hsb,
                                              float* __restrict__ cf,
                                              float* __restrict__ cb, int t) {
    __shared__ uint4 ldsB4[64 * ST_LDSTRIDE / 16];
    char* ldsB = (char*)ldsB4;
    const int dir = blockIdx.y;
    const int l = dir ? (L_ - 1 - t) : t;
    const __bf16* hprev = dir ? hsb + (size_t)(l + 1) * B_ * H_
                              : hsf + (size_t)t * B_ * H_;
    __bf16* hout        = dir ? hsb + (size_t)l * B_ * H_
                              : hsf + (size_t)(t + 1) * B_ * H_;
    const float* xp = (dir ? xpb : xpf) + (size_t)l * B_ * G4H;
    float* cst = dir ? cb : cf;
    const __bf16* Wh = dir ? Whb : Whf;

    const int lane = threadIdx.x & 31, wave = threadIdx.x >> 5;
    const int jg = blockIdx.x;                 // 0..63 -> 16 h-cols
    const int mr = wave * 16 + (lane & 15);    // batch row for A
    const int nc = lane & 15;

#if USE_TDM
    {
        int g = wave;                          // one gate chunk per wave
        tdm_load_2d_pad(Wh + (size_t)(g * H_ + jg * 16) * H_,
                        lds_off32(ldsB) + (unsigned)(g * 16 * ST_LDSTRIDE),
                        H_, 16, H_);
        __builtin_amdgcn_s_wait_tensorcnt(0);
    }
#else
    for (int u = threadIdx.x; u < 64 * 128; u += 128) {  // 16B units: 64 rows x 128
        int r = u >> 7, kdw = (u & 127) * 4;
        int g = r >> 4, rr = r & 15;
        *(uint4*)(ldsB + r * ST_LDSTRIDE + kdw * 4 + (kdw >= 256 ? 16 : 0)) =
            *(const uint4*)(Wh + (size_t)(g * H_ + jg * 16 + rr) * H_ + kdw * 2);
    }
#endif
    __syncthreads();

    v8f c0 = {}, c1 = {}, c2 = {}, c3 = {};
    for (int k0 = 0; k0 < H_; k0 += 32) {
        v16bf a  = load_a(hprev, H_, mr, k0, lane);
        v16bf b0 = load_b_ldsp(ldsB,  0 + nc, k0, lane, ST_LDSTRIDE);
        v16bf b1 = load_b_ldsp(ldsB, 16 + nc, k0, lane, ST_LDSTRIDE);
        v16bf b2 = load_b_ldsp(ldsB, 32 + nc, k0, lane, ST_LDSTRIDE);
        v16bf b3 = load_b_ldsp(ldsB, 48 + nc, k0, lane, ST_LDSTRIDE);
        c0 = wmma_bf16(a, b0, c0);
        c1 = wmma_bf16(a, b1, c1);
        c2 = wmma_bf16(a, b2, c2);
        c3 = wmma_bf16(a, b3, c3);
    }
    const int hi = lane >> 4;
    const int hcol = jg * 16 + nc;
    for (int r = 0; r < 8; r++) {
        int b = wave * 16 + r + 8 * hi;
        const float* xr = xp + (size_t)b * G4H;
        float gi = c0[r] + xr[0 * H_ + hcol];
        float gf = c1[r] + xr[1 * H_ + hcol];
        float gg = c2[r] + xr[2 * H_ + hcol];
        float go = c3[r] + xr[3 * H_ + hcol];
        float cp = cst[b * H_ + hcol];
        float si = 1.f / (1.f + __expf(-gi));
        float sf = 1.f / (1.f + __expf(-gf));
        float so = 1.f / (1.f + __expf(-go));
        float cn = sf * cp + si * tanhf(gg);
        cst[b * H_ + hcol] = cn;
        hout[b * H_ + hcol] = f2bf(so * tanhf(cn));
    }
}

// ---- emissions = [h_f ; h_b] @ W_out^T + b_out : M=16384, N=32(pad), K=2048
__global__ void __launch_bounds__(256) k_emis(const __bf16* __restrict__ hsf,
                                              const __bf16* __restrict__ hsb,
                                              const __bf16* __restrict__ Wo,
                                              const float* __restrict__ bout,
                                              float* __restrict__ emis) {
    const int lane = threadIdx.x & 31, wave = threadIdx.x >> 5;
    const int m0 = blockIdx.x * 128 + wave * 16;
    const int mr = m0 + (lane & 15);
    const int lrow = mr >> 6, brow = mr & 63;   // 16 | B, so l const per row tile
    const int nc = lane & 15;
    v8f c0 = {}, c1 = {};
    const __bf16* Af = hsf + ((size_t)(lrow + 1) * B_ + brow) * H_;
    for (int k0 = 0; k0 < H_; k0 += 32) {
        v16bf a  = load_a_row(Af, k0, lane);
        v16bf b0 = load_b(Wo, K2H, nc,      k0, lane);
        v16bf b1 = load_b(Wo, K2H, 16 + nc, k0, lane);
        c0 = wmma_bf16(a, b0, c0);
        c1 = wmma_bf16(a, b1, c1);
    }
    const __bf16* Ab = hsb + ((size_t)lrow * B_ + brow) * H_;
    for (int k0 = 0; k0 < H_; k0 += 32) {
        v16bf a  = load_a_row(Ab, k0, lane);
        v16bf b0 = load_b(Wo, K2H, nc,      H_ + k0, lane);
        v16bf b1 = load_b(Wo, K2H, 16 + nc, H_ + k0, lane);
        c0 = wmma_bf16(a, b0, c0);
        c1 = wmma_bf16(a, b1, c1);
    }
    const int hi = lane >> 4;
    float bv0 = (nc < T_) ? bout[nc] : 0.f;
    float bv1 = (16 + nc < T_) ? bout[16 + nc] : 0.f;
    for (int r = 0; r < 8; r++) {
        int m = m0 + r + 8 * hi;
        float* o = emis + (size_t)m * T_;
        if (nc < T_)      o[nc]      = c0[r] + bv0;
        if (16 + nc < T_) o[16 + nc] = c1[r] + bv1;
    }
}

// ---- CRF scan: one block (32 lanes) per batch sequence --------------------
__global__ void __launch_bounds__(32) k_crf(const float* __restrict__ emis,
                                            const int* __restrict__ tags,
                                            const unsigned char* __restrict__ mask,
                                            const float* __restrict__ startT,
                                            const float* __restrict__ endT,
                                            const float* __restrict__ trans,
                                            float* __restrict__ val) {
    const int b = blockIdx.x, lane = threadIdx.x;
    __shared__ float sc[32];
    __shared__ float tr[T_ * T_];
    __shared__ float red[32];
    for (int i = lane; i < T_ * T_; i += 32) tr[i] = trans[i];
    float s = -1e30f;
    if (lane < T_) s = startT[lane] + emis[(size_t)(0 * B_ + b) * T_ + lane];
    sc[lane] = s;
    __syncthreads();
    for (int l = 1; l < L_; l++) {
        unsigned char mm = mask[b * L_ + l];
        if (lane < T_) {
            float em = emis[(size_t)(l * B_ + b) * T_ + lane];
            float mx = -1e30f;
            for (int p = 0; p < T_; p++) mx = fmaxf(mx, sc[p] + tr[p * T_ + lane]);
            float sm = 0.f;
            for (int p = 0; p < T_; p++) sm += __expf(sc[p] + tr[p * T_ + lane] - mx);
            float nxt = mx + __logf(sm) + em;
            s = mm ? nxt : s;
        }
        __syncthreads();
        sc[lane] = s;
        __syncthreads();
    }
    red[lane] = (lane < T_) ? s + endT[lane] : -1e30f;
    __syncthreads();
    float mx = -1e30f;
    for (int p = 0; p < T_; p++) mx = fmaxf(mx, red[p]);
    float sm = 0.f;
    for (int p = 0; p < T_; p++) sm += __expf(red[p] - mx);
    float logZ = mx + __logf(sm);
    float part = 0.f;
    for (int l = 1 + lane; l < L_; l += 32) {
        int tp = tags[b * L_ + l - 1], tc = tags[b * L_ + l];
        float mm = mask[b * L_ + l] ? 1.f : 0.f;
        part += (tr[tp * T_ + tc] + emis[(size_t)(l * B_ + b) * T_ + tc]) * mm;
    }
    red[lane] = part;
    __syncthreads();
    if (lane == 0) {
        float num = 0.f;
        for (int p = 0; p < 32; p++) num += red[p];
        int t0 = tags[b * L_];
        num += startT[t0] + emis[(size_t)(0 * B_ + b) * T_ + t0];
        int cnt = 0;
        for (int l = 0; l < L_; l++) cnt += mask[b * L_ + l] ? 1 : 0;
        num += endT[tags[b * L_ + (cnt - 1)]];
        val[b] = num - logZ;
    }
}

__global__ void __launch_bounds__(64) k_loss(const float* __restrict__ val,
                                             float* __restrict__ out) {
    __shared__ float r[64];
    r[threadIdx.x] = val[threadIdx.x];
    __syncthreads();
    if (threadIdx.x == 0) {
        float s = 0.f;
        for (int j = 0; j < B_; j++) s += r[j];
        out[0] = -s / (float)B_;
    }
}

// ---------------------------------------------------------------------------
extern "C" void kernel_launch(void* const* d_in, const int* in_sizes, int n_in,
                              void* d_out, int out_size, void* d_ws, size_t ws_size,
                              hipStream_t stream) {
    (void)in_sizes; (void)n_in; (void)out_size; (void)ws_size;
    const int*   input_ids = (const int*)d_in[0];
    const int*   tags      = (const int*)d_in[1];
    const unsigned char* mask = (const unsigned char*)d_in[2];  // jnp bool = 1B
    const float* embed     = (const float*)d_in[3];
    const float* WihF      = (const float*)d_in[4];
    const float* WhhF      = (const float*)d_in[5];
    const float* bihF      = (const float*)d_in[6];
    const float* bhhF      = (const float*)d_in[7];
    const float* WihB      = (const float*)d_in[8];
    const float* WhhB      = (const float*)d_in[9];
    const float* bihB      = (const float*)d_in[10];
    const float* bhhB      = (const float*)d_in[11];
    const float* Wout      = (const float*)d_in[12];
    const float* bout      = (const float*)d_in[13];
    const float* startT    = (const float*)d_in[14];
    const float* endT      = (const float*)d_in[15];
    const float* trans     = (const float*)d_in[16];
    float* out = (float*)d_out;

    char* ws = (char*)d_ws;
    size_t off = 0;
    auto take = [&](size_t bytes) {
        char* p = ws + off;
        off = (off + bytes + 255) & ~(size_t)255;
        return p;
    };
    __bf16* WihF_bf = (__bf16*)take((size_t)G4H * E_ * 2);
    __bf16* WihB_bf = (__bf16*)take((size_t)G4H * E_ * 2);
    __bf16* WhhF_bf = (__bf16*)take((size_t)G4H * H_ * 2);
    __bf16* WhhB_bf = (__bf16*)take((size_t)G4H * H_ * 2);
    __bf16* Wout_bf = (__bf16*)take((size_t)32 * K2H * 2);
    __bf16* xbf     = (__bf16*)take((size_t)M1 * E_ * 2);
    float*  bsf     = (float*)take((size_t)G4H * 4);
    float*  bsb     = (float*)take((size_t)G4H * 4);
    float*  xpf     = (float*)take((size_t)M1 * G4H * 4);   // 256 MB
    float*  xpb     = (float*)take((size_t)M1 * G4H * 4);   // 256 MB
    __bf16* hsf     = (__bf16*)take((size_t)(L_ + 1) * B_ * H_ * 2);
    __bf16* hsb     = (__bf16*)take((size_t)(L_ + 1) * B_ * H_ * 2);
    float*  cf      = (float*)take((size_t)B_ * H_ * 4);
    float*  cb      = (float*)take((size_t)B_ * H_ * 4);
    float*  emis    = (float*)take((size_t)M1 * T_ * 4);
    float*  val     = (float*)take((size_t)B_ * 4);

    // setup / conversion
    k_init<<<(B_ * H_ + 255) / 256, 256, 0, stream>>>(hsf, hsb + (size_t)L_ * B_ * H_, cf, cb);
    {
        int n = G4H * E_;
        k_cvt_bf16<<<(n + 255) / 256, 256, 0, stream>>>(WihF, WihF_bf, n);
        k_cvt_bf16<<<(n + 255) / 256, 256, 0, stream>>>(WihB, WihB_bf, n);
        n = G4H * H_;
        k_cvt_bf16<<<(n + 255) / 256, 256, 0, stream>>>(WhhF, WhhF_bf, n);
        k_cvt_bf16<<<(n + 255) / 256, 256, 0, stream>>>(WhhB, WhhB_bf, n);
    }
    k_pack_wout<<<(32 * K2H + 255) / 256, 256, 0, stream>>>(Wout, Wout_bf);
    k_bias<<<(G4H + 255) / 256, 256, 0, stream>>>(bihF, bhhF, bihB, bhhB, bsf, bsb);
    k_embed<<<M1, 256, 0, stream>>>(input_ids, embed, xbf);

    // input projection GEMM (both directions)
    k_xproj<<<dim3(M1 / 128, G4H / 64, 2), 256, 0, stream>>>(
        xbf, WihF_bf, WihB_bf, bsf, bsb, xpf, xpb);

    // sequential recurrence: fwd + bwd in parallel per step
    for (int t = 0; t < L_; t++)
        k_step<<<dim3(H_ / 16, 2), 128, 0, stream>>>(
            xpf, xpb, WhhF_bf, WhhB_bf, hsf, hsb, cf, cb, t);

    // emissions GEMM
    k_emis<<<M1 / 128, 256, 0, stream>>>(hsf, hsb, Wout_bf, bout, emis);

    // CRF per sequence + final loss
    k_crf<<<B_, 32, 0, stream>>>(emis, tags, mask, startT, endT, trans, val);
    k_loss<<<1, 64, 0, stream>>>(val, out);
}
